// SparseMoE_8100308320512
// MI455X (gfx1250) — compile-verified, matrix operand order checked by reading, not run on hardware
//
#include <hip/hip_runtime.h>
#include <hip/hip_bf16.h>

// ---------------- problem constants ----------------
constexpr int BB = 4, SS = 4096, DD = 256, EE = 8, HH = 512;
constexpr int NTOK = BB * SS;          // 16384 tokens
constexpr int MT   = 128;              // token tile (rows per FFN block)
constexpr float LN_EPS = 1e-5f;

// ---------------- WMMA types ----------------
typedef __attribute__((ext_vector_type(16))) __bf16 v16bf;
typedef __attribute__((ext_vector_type(8)))  float  v8f;
typedef __attribute__((ext_vector_type(4)))  int    v4i;

union AF { v16bf v; uint4 q[2]; };     // 16x32 bf16 A/B fragment = 32 bytes/lane

// ---------------- async global->LDS copy (CDNA5) ----------------
// Probe-confirmed (round 2 diagnostic): builtin exists, 4 args, arg0 is
// 'int __vector(4) __device__*' (global int4*). Cast through integers to
// avoid invalid cross-AS pointer casts.
#if defined(__AMDGCN__)
#if __has_builtin(__builtin_amdgcn_global_load_async_to_lds_b128)
#define HAVE_ASYNC_LDS 1
typedef __attribute__((address_space(1))) v4i* g_v4i_ptr;
typedef __attribute__((address_space(3))) v4i* l_v4i_ptr;
#define ASYNC_CP16(dst, src)                                                   \
  __builtin_amdgcn_global_load_async_to_lds_b128(                              \
      (g_v4i_ptr)(unsigned long long)(uintptr_t)(src),                         \
      (l_v4i_ptr)(unsigned)(uintptr_t)(dst), 0, 0)
#endif
#endif

__device__ __forceinline__ void copy16B_g2l(__bf16* dst, const __bf16* src) {
#ifdef HAVE_ASYNC_LDS
  ASYNC_CP16(dst, src);
#else
  *(uint4*)dst = *(const uint4*)src;
#endif
}

__device__ __forceinline__ void wait_async_copies() {
#ifdef HAVE_ASYNC_LDS
#if __has_builtin(__builtin_amdgcn_s_wait_asynccnt)
  __builtin_amdgcn_s_wait_asynccnt(0);
#else
  asm volatile("s_wait_asynccnt 0x0" ::: "memory");
#endif
#endif
}

// ---------------- LDS tile strides (bf16 elems, +8 pad vs bank conflicts) ----
constexpr int XS_STRIDE  = 256 + 8;    // x tile  [128 rows][K=256]
constexpr int W1S_STRIDE = 256 + 8;    // W1 tile [N=128 (h)][K=256 (d)]
constexpr int W2S_STRIDE = 128 + 8;    // W2 tile [N=128 (d)][K=128 (h)]
constexpr int HS_STRIDE  = 512 + 8;    // h tile  [128 rows][K=512]
constexpr int XS_ELEMS = 128 * XS_STRIDE;
constexpr int WS_ELEMS = 128 * W1S_STRIDE;   // shared region reused for W1/W2 tiles
constexpr int HS_ELEMS = 128 * HS_STRIDE;
constexpr size_t SMEM_BYTES =
    (size_t)(XS_ELEMS + WS_ELEMS + HS_ELEMS) * 2 + 128 * 4 + 128 * 4;

// ============================================================
// Kernel 0: zero per-expert routed-token counters
// ============================================================
__global__ void moe_init_counts(int* counts) {
  if (threadIdx.x < EE) counts[threadIdx.x] = 0;
}

// ============================================================
// Kernel 1: transpose + fp32->bf16 convert of expert weights
//   W1t[e][h][d] = bf16(W1[e][d][h])   (K=d contiguous -> B-frag friendly)
//   W2t[e][d][h] = bf16(W2[e][h][d])   (K=h contiguous -> B-frag friendly)
// ============================================================
__global__ void moe_convert_weights(const float* __restrict__ W1,
                                    const float* __restrict__ W2,
                                    __bf16* __restrict__ W1t,
                                    __bf16* __restrict__ W2t) {
  constexpr int TOT = EE * DD * HH;    // 1,048,576 per tensor
  int o = blockIdx.x * blockDim.x + threadIdx.x;
  if (o < TOT) {
    int d = o & (DD - 1);
    int h = (o >> 8) & (HH - 1);
    int e = o >> 17;
    W1t[o] = (__bf16)W1[((size_t)e * DD + d) * HH + h];
  } else {
    int p = o - TOT;
    if (p < TOT) {
      int h = p & (HH - 1);
      int d = (p >> 9) & (DD - 1);
      int e = p >> 17;
      W2t[p] = (__bf16)W2[((size_t)e * HH + h) * DD + d];
    }
  }
}

// ============================================================
// Kernel 2: gating/routing. One wave (32 lanes) per token.
// ============================================================
__global__ __launch_bounds__(256) void moe_route(
    const float* __restrict__ x, const float* __restrict__ Wg,
    const float* __restrict__ bg, const float* __restrict__ b2,
    int* __restrict__ counts, int* __restrict__ tokIdx,
    float* __restrict__ tokW, float* __restrict__ accBuf) {
  int wid  = threadIdx.x >> 5;
  int lane = threadIdx.x & 31;
  int t = blockIdx.x * 8 + wid;
  if (t >= NTOK) return;

  const float* xr = x + (size_t)t * DD + lane * 8;
  float p[EE];
#pragma unroll
  for (int e = 0; e < EE; ++e) p[e] = 0.f;
#pragma unroll
  for (int j = 0; j < 8; ++j) {
    float xv = xr[j];
    const float4* wr = (const float4*)(Wg + (size_t)(lane * 8 + j) * EE);
    float4 w0 = wr[0], w1 = wr[1];
    p[0] += xv * w0.x; p[1] += xv * w0.y; p[2] += xv * w0.z; p[3] += xv * w0.w;
    p[4] += xv * w1.x; p[5] += xv * w1.y; p[6] += xv * w1.z; p[7] += xv * w1.w;
  }
#pragma unroll
  for (int e = 0; e < EE; ++e) {
#pragma unroll
    for (int off = 16; off > 0; off >>= 1) p[e] += __shfl_xor(p[e], off, 32);
    p[e] += bg[e];
  }
  int e0 = 0; float l0 = p[0];
#pragma unroll
  for (int e = 1; e < EE; ++e) if (p[e] > l0) { l0 = p[e]; e0 = e; }
  int e1 = (e0 == 0) ? 1 : 0; float l1 = p[e1];
#pragma unroll
  for (int e = 0; e < EE; ++e)
    if (e != e0 && p[e] > l1) { l1 = p[e]; e1 = e; }
  float a1 = __expf(l1 - l0);
  float inv = 1.0f / (1.0f + a1);
  float w0 = inv, w1 = a1 * inv;

  if (lane == 0) {
    int p0 = atomicAdd(&counts[e0], 1);
    tokIdx[e0 * NTOK + p0] = t; tokW[e0 * NTOK + p0] = w0;
    int p1 = atomicAdd(&counts[e1], 1);
    tokIdx[e1 * NTOK + p1] = t; tokW[e1 * NTOK + p1] = w1;
  }
  float* ar = accBuf + (size_t)t * DD + lane * 8;
  const float* b2a = b2 + (size_t)e0 * DD + lane * 8;
  const float* b2b = b2 + (size_t)e1 * DD + lane * 8;
#pragma unroll
  for (int j = 0; j < 8; ++j) ar[j] = xr[j] + w0 * b2a[j] + w1 * b2b[j];
}

// ============================================================
// Kernel 3: expert FFN, WMMA bf16, async LDS staging for weights.
// grid = (NTOK/128 tiles, E experts); 256 threads = 8 waves.
// ============================================================
__global__ __launch_bounds__(256) void moe_ffn(
    const float* __restrict__ x, const float* __restrict__ b1,
    const int* __restrict__ counts, const int* __restrict__ tokIdx,
    const float* __restrict__ tokW, const __bf16* __restrict__ W1t,
    const __bf16* __restrict__ W2t, float* __restrict__ accBuf) {
  extern __shared__ __align__(16) char smem[];
  __bf16* Xs = (__bf16*)smem;          // [128][XS_STRIDE]  gathered x rows (bf16)
  __bf16* Ws = Xs + XS_ELEMS;          // W1 slice / W2 slice (reused)
  __bf16* Hs = Ws + WS_ELEMS;          // [128][HS_STRIDE]  relu(h) rows (bf16)
  int*   sTok = (int*)(Hs + HS_ELEMS); // [128] token indices
  float* sW   = (float*)(sTok + 128);  // [128] gate weights

  const int e = blockIdx.y, tile = blockIdx.x;
  const int cnt = counts[e];
  const int base = tile * MT;
  if (base >= cnt) return;

  const int tid = threadIdx.x, lane = tid & 31, wv = tid >> 5;
  const int lm = lane & 15;            // column / M-row within tile
  const int lh = lane >> 4;            // K-half select
  const int m0 = wv * 16;              // this wave's 16-token row block

  if (tid < MT) {
    int g = base + tid;
    if (g < cnt) { sTok[tid] = tokIdx[e * NTOK + g]; sW[tid] = tokW[e * NTOK + g]; }
    else         { sTok[tid] = 0;                    sW[tid] = 0.f; }
  }
  __syncthreads();

  // gather 128 x rows -> bf16 LDS (each thread: half a row; converts, so sync path)
  {
    int r = tid >> 1, half = tid & 1;
    const float* src = x + (size_t)sTok[r] * DD + half * 128;
    __bf16* dst = Xs + r * XS_STRIDE + half * 128;
#pragma unroll
    for (int i = 0; i < 32; ++i) {
      float4 v = ((const float4*)src)[i];
      union { __bf16 h[4]; uint2 u; } t;
      t.h[0] = (__bf16)v.x; t.h[1] = (__bf16)v.y;
      t.h[2] = (__bf16)v.z; t.h[3] = (__bf16)v.w;
      ((uint2*)dst)[i] = t.u;
    }
  }

  // ---------------- GEMM1: h = relu(Xs * W1[e] + b1[e]) ----------------
  for (int hs = 0; hs < 4; ++hs) {
    __syncthreads();   // prior compute done (and Xs staged, on hs==0)
    {                  // stage W1 slice async: [n=h'][k=d], rows K-contiguous
      int n = tid >> 1, half = tid & 1;
      const __bf16* srcw = W1t + ((size_t)e * HH + hs * 128 + n) * DD + half * 128;
      __bf16* dstw = Ws + n * W1S_STRIDE + half * 128;
      __builtin_prefetch(srcw + 128 * DD, 0, 0);   // next slice
#pragma unroll
      for (int i = 0; i < 16; ++i) copy16B_g2l(dstw + i * 8, srcw + i * 8);
      wait_async_copies();
    }
    __syncthreads();

    v8f acc[8];
#pragma unroll
    for (int n = 0; n < 8; ++n) acc[n] = v8f{};
#pragma unroll
    for (int kk = 0; kk < 8; ++kk) {               // K = 256 = 8 * 32
      AF a;
      const __bf16* pa = Xs + (m0 + lm) * XS_STRIDE + kk * 32 + lh * 8;
      a.q[0] = *(const uint4*)pa;
      a.q[1] = *(const uint4*)(pa + 16);
#pragma unroll
      for (int n = 0; n < 8; ++n) {
        AF b;
        const __bf16* pb = Ws + (n * 16 + lm) * W1S_STRIDE + kk * 32 + lh * 8;
        b.q[0] = *(const uint4*)pb;
        b.q[1] = *(const uint4*)(pb + 16);
        acc[n] = __builtin_amdgcn_wmma_f32_16x16x32_bf16(
            false, a.v, false, b.v, (short)0, acc[n], false, false);
      }
    }
#pragma unroll
    for (int n = 0; n < 8; ++n) {
      int col = hs * 128 + n * 16 + lm;
      float bias = b1[e * HH + col];
#pragma unroll
      for (int r = 0; r < 8; ++r) {
        int row = m0 + lh * 8 + r;
        float v = acc[n][r] + bias;
        v = v > 0.f ? v : 0.f;
        Hs[row * HS_STRIDE + col] = (__bf16)v;
      }
    }
  }

  // ---------------- GEMM2: y = Hs * W2[e]; acc += w * y (atomic) ----------
  for (int half = 0; half < 2; ++half) {
    v8f acc2[8];
#pragma unroll
    for (int n = 0; n < 8; ++n) acc2[n] = v8f{};
    for (int ks = 0; ks < 4; ++ks) {               // K = 512 = 4 * 128 slices
      __syncthreads();
      {                // stage W2 slice async: [n=d][k=h], rows K-contiguous
        int n = tid >> 1, hh = tid & 1;
        const __bf16* srcw =
            W2t + ((size_t)e * DD + half * 128 + n) * HH + ks * 128 + hh * 64;
        __bf16* dstw = Ws + n * W2S_STRIDE + hh * 64;
#pragma unroll
        for (int i = 0; i < 8; ++i) copy16B_g2l(dstw + i * 8, srcw + i * 8);
        wait_async_copies();
      }
      __syncthreads();
#pragma unroll
      for (int kk = 0; kk < 4; ++kk) {             // 128 = 4 * 32
        AF a;
        const __bf16* pa =
            Hs + (m0 + lm) * HS_STRIDE + ks * 128 + kk * 32 + lh * 8;
        a.q[0] = *(const uint4*)pa;
        a.q[1] = *(const uint4*)(pa + 16);
#pragma unroll
        for (int n = 0; n < 8; ++n) {
          AF b;
          const __bf16* pb = Ws + (n * 16 + lm) * W2S_STRIDE + kk * 32 + lh * 8;
          b.q[0] = *(const uint4*)pb;
          b.q[1] = *(const uint4*)(pb + 16);
          acc2[n] = __builtin_amdgcn_wmma_f32_16x16x32_bf16(
              false, a.v, false, b.v, (short)0, acc2[n], false, false);
        }
      }
    }
#pragma unroll
    for (int n = 0; n < 8; ++n) {
      int col = half * 128 + n * 16 + lm;
#pragma unroll
      for (int r = 0; r < 8; ++r) {
        int row = m0 + lh * 8 + r;
        float v = sW[row] * acc2[n][r];
        atomicAdd(accBuf + (size_t)sTok[row] * DD + col, v);
      }
    }
  }
}

// ============================================================
// Kernel 4: LayerNorm(acc) -> out. One wave per token.
// ============================================================
__global__ __launch_bounds__(256) void moe_layernorm(
    const float* __restrict__ accBuf, const float* __restrict__ gamma,
    const float* __restrict__ beta, float* __restrict__ out) {
  int wid  = threadIdx.x >> 5;
  int lane = threadIdx.x & 31;
  int t = blockIdx.x * 8 + wid;
  if (t >= NTOK) return;

  const float* zr = accBuf + (size_t)t * DD + lane * 8;
  float z[8];
  float s = 0.f;
#pragma unroll
  for (int j = 0; j < 8; ++j) { z[j] = zr[j]; s += z[j]; }
#pragma unroll
  for (int off = 16; off > 0; off >>= 1) s += __shfl_xor(s, off, 32);
  float mu = s * (1.0f / DD);
  float vs = 0.f;
#pragma unroll
  for (int j = 0; j < 8; ++j) { float d = z[j] - mu; vs += d * d; }
#pragma unroll
  for (int off = 16; off > 0; off >>= 1) vs += __shfl_xor(vs, off, 32);
  float inv = rsqrtf(vs * (1.0f / DD) + LN_EPS);

  float* orow = out + (size_t)t * DD + lane * 8;
  const float* gr = gamma + lane * 8;
  const float* br = beta + lane * 8;
#pragma unroll
  for (int j = 0; j < 8; ++j) orow[j] = gr[j] * (z[j] - mu) * inv + br[j];
}

// ============================================================
// host launch
// ============================================================
extern "C" void kernel_launch(void* const* d_in, const int* in_sizes, int n_in,
                              void* d_out, int out_size, void* d_ws,
                              size_t ws_size, hipStream_t stream) {
  (void)in_sizes; (void)n_in; (void)out_size; (void)ws_size;
  const float* x     = (const float*)d_in[0];
  const float* Wg    = (const float*)d_in[1];
  const float* bg    = (const float*)d_in[2];
  const float* W1    = (const float*)d_in[3];
  const float* b1    = (const float*)d_in[4];
  const float* W2    = (const float*)d_in[5];
  const float* b2    = (const float*)d_in[6];
  const float* gamma = (const float*)d_in[7];
  const float* beta  = (const float*)d_in[8];
  float* out = (float*)d_out;

  char* ws = (char*)d_ws;
  int*    counts = (int*)ws;
  int*    tokIdx = (int*)(ws + 256);
  float*  tokW   = (float*)(ws + 256 + (size_t)EE * NTOK * 4);
  float*  accBuf = (float*)(ws + 256 + (size_t)2 * EE * NTOK * 4);
  __bf16* W1t = (__bf16*)(ws + 256 + (size_t)2 * EE * NTOK * 4 +
                          (size_t)NTOK * DD * 4);
  __bf16* W2t = W1t + (size_t)EE * HH * DD;

  (void)hipFuncSetAttribute((const void*)moe_ffn,
                            hipFuncAttributeMaxDynamicSharedMemorySize,
                            (int)SMEM_BYTES);

  moe_init_counts<<<1, 32, 0, stream>>>(counts);
  moe_convert_weights<<<2 * (EE * DD * HH) / 256, 256, 0, stream>>>(W1, W2,
                                                                    W1t, W2t);
  moe_route<<<NTOK / 8, 256, 0, stream>>>(x, Wg, bg, b2, counts, tokIdx, tokW,
                                          accBuf);
  moe_ffn<<<dim3(NTOK / MT, EE), 256, SMEM_BYTES, stream>>>(
      x, b1, counts, tokIdx, tokW, W1t, W2t, accBuf);
  moe_layernorm<<<NTOK / 8, 256, 0, stream>>>(accBuf, gamma, beta, out);
}
